// NNUEBucketModel_33389075759704
// MI455X (gfx1250) — matrix operand order repaired and checked by reading, not compiled, not run on hardware
//
#include <hip/hip_runtime.h>
#include <hip/hip_bf16.h>

typedef __bf16 bf16_t;
typedef __attribute__((ext_vector_type(16))) __bf16 v16bf;
typedef __attribute__((ext_vector_type(8)))  __bf16 v8bf;
typedef __attribute__((ext_vector_type(8)))  float  v8f;
typedef __attribute__((ext_vector_type(4)))  float  v4f;

#define IN_FEATS 770
#define FT       512
#define KPAD     800     // 25 chunks of 32
#define NCHUNK   25
#define COMB     1024
#define H1OUT    256     // 8 buckets * 32

#define XROW     36      // padded f32 row stride for x tile
#define XBUF     (32 * XROW)

// workspace byte offsets (all 32B aligned)
#define WT_OFF   0                         // 800*512*2  = 819200
#define H1T_OFF  819200                    // 1024*256*2 = 524288
#define H2T_OFF  (819200 + 524288)         // 8*32*32*2  = 16384

// Build a 16-bit WMMA A-fragment from two 16-byte LDS chunks (bf16 source).
// Layout (ISA 7.12.2): lane = 16h+m holds row m, K = {8h..8h+7} and {16+8h..23+8h}.
__device__ __forceinline__ v16bf make_afrag(const bf16_t* lo_p, const bf16_t* hi_p) {
    v8bf lo = *(const v8bf*)lo_p;
    v8bf hi = *(const v8bf*)hi_p;
    v16bf a;
#pragma unroll
    for (int i = 0; i < 8; ++i) { a[i] = lo[i]; a[i + 8] = hi[i]; }
    return a;
}

// Same, but from an f32 LDS tile with on-the-fly bf16 conversion
// (v_cvt_pk_bf16_f32 co-executes with the WMMA/XDL pipe).
__device__ __forceinline__ v16bf make_afrag_f32(const float* lo_p, const float* hi_p) {
    v4f a0 = *(const v4f*)(lo_p + 0);
    v4f a1 = *(const v4f*)(lo_p + 4);
    v4f b0 = *(const v4f*)(hi_p + 0);
    v4f b1 = *(const v4f*)(hi_p + 4);
    v16bf a;
#pragma unroll
    for (int i = 0; i < 4; ++i) {
        a[i]      = (bf16_t)a0[i];
        a[i + 4]  = (bf16_t)a1[i];
        a[i + 8]  = (bf16_t)b0[i];
        a[i + 12] = (bf16_t)b1[i];
    }
    return a;
}

// ---------------------------------------------------------------------------
// Setup: transpose/convert weights to bf16 K-major layouts for B-fragments.
// ---------------------------------------------------------------------------
__global__ void convert_weights(const float* __restrict__ ftw,
                                const float* __restrict__ h1w,
                                const float* __restrict__ h2w,
                                bf16_t* __restrict__ wT,
                                bf16_t* __restrict__ h1t,
                                bf16_t* __restrict__ h2t) {
    int idx = blockIdx.x * 256 + threadIdx.x;
    if (idx < KPAD * FT) {
        int n = idx / KPAD, k = idx % KPAD;
        float v = (k < IN_FEATS) ? ftw[n * IN_FEATS + k] : 0.0f;
        wT[idx] = (bf16_t)v;
    }
    if (idx < COMB * H1OUT) {
        int n = idx / COMB, k = idx % COMB;
        h1t[idx] = (bf16_t)h1w[(n >> 5) * 32 * COMB + (n & 31) * COMB + k];
    }
    if (idx < 8 * 32 * 32) {
        int bk = idx >> 10, o = (idx >> 5) & 31, p = idx & 31;
        h2t[idx] = (bf16_t)h2w[bk * 1024 + o * 32 + p];
    }
}

// ---------------------------------------------------------------------------
// Fused NNUE forward: 16 samples per 256-thread block (8 wave32).
// Virtual rows: r in [0,32): sample = r&15, side = r>>4 (0=stm, 1=nstm).
// x tile is streamed into LDS with GLOBAL_LOAD_ASYNC_TO_LDS_B64 (ASYNCcnt),
// double-buffered so chunk c+1 streams while chunk c's WMMAs execute.
// ---------------------------------------------------------------------------
__global__ __launch_bounds__(256) void nnue_fused(
    const float*  __restrict__ X,
    const float*  __restrict__ ft_b,
    const float*  __restrict__ h1_b,
    const float*  __restrict__ h2_b,
    const float*  __restrict__ h3_w,
    const float*  __restrict__ h3_b,
    const bf16_t* __restrict__ wT,
    const bf16_t* __restrict__ h1t,
    const bf16_t* __restrict__ h2t,
    float*        __restrict__ out) {
    __shared__ __align__(16) float  ldsXF[2 * XBUF];   // x tile f32, double buffered
    __shared__ __align__(16) float  ldsTrash[4];       // async sink for OOB pieces
    __shared__ __align__(16) bf16_t ldsC[16 * 1032];   // combined: 16 x 1024 (pad->1032)
    __shared__ __align__(16) bf16_t ldsH[16 * 264];    // h1 out: 16 x 256 (pad->264)
    __shared__ __align__(16) bf16_t ldsH2[16 * 264];   // h2 out
    __shared__ int ldsBk[16];                          // bucket per sample

    const int t    = threadIdx.x;
    const int lane = t & 31;
    const int w    = t >> 5;          // wave id 0..7
    const int hh   = lane >> 4;       // half-wave
    const int m    = lane & 15;
    const int b0   = blockIdx.x * 16;

    // staging role: 8 threads per row, 4 floats (two b64 pieces) each
    const int sr     = t >> 3;        // virtual row 0..31
    const int sj     = (t & 7) * 4;   // K sub-offset within chunk
    const int s_side = sr >> 4;
    const int s_smp  = sr & 15;
    const float* xr = X + (size_t)(b0 + s_smp) * (2 * IN_FEATS) + s_side * IN_FEATS;

    // FT tiling: wave -> row-tile (2) x col-group (4 groups of 128 cols)
    const int rt = w >> 2;
    const int cg = w & 3;

    const unsigned trashAddr = (unsigned)(uintptr_t)&ldsTrash[0];

    // Issue one chunk's x tile as async global->LDS copies (2 x b64 per thread;
    // uniform ASYNCcnt: invalid tail pieces are redirected to a trash slot and
    // their LDS slots zero-filled with ordinary (compiler-tracked) stores).
    auto issue_chunk = [&](int c) {
        float* bufp = &ldsXF[(c & 1) * XBUF];
        const int k0 = c * 32;
#pragma unroll
        for (int piece = 0; piece < 2; ++piece) {
            const int p = k0 + sj + piece * 2;          // even float index in row
            const bool valid = (p + 1) < IN_FEATS;
            float* dstp = &bufp[sr * XROW + sj + piece * 2];
            if (!valid) { dstp[0] = 0.0f; dstp[1] = 0.0f; }
            const unsigned ldsaddr = valid ? (unsigned)(uintptr_t)dstp : trashAddr;
            const float* gp = valid ? (xr + p) : xr;    // dummy src stays in-bounds
            asm volatile("global_load_async_to_lds_b64 %0, %1, off"
                         :: "v"(ldsaddr),
                            "v"((unsigned long long)(uintptr_t)gp)
                         : "memory");
        }
        if (c + 2 < NCHUNK) __builtin_prefetch(xr + (c + 2) * 32 + sj, 0, 0);
    };

    const v8f zero8 = {0.f, 0.f, 0.f, 0.f, 0.f, 0.f, 0.f, 0.f};
    v8f acc[8];
#pragma unroll
    for (int i = 0; i < 8; ++i) acc[i] = zero8;

    float psum = 0.0f;   // piece-count partial (side 0, k<768 only)

    // ---------------- FT layer: combined = crelu(x @ ft_w^T + ft_b) ----------
    issue_chunk(0);
    for (int c = 0; c < NCHUNK; ++c) {
        if (c + 1 < NCHUNK) {
            issue_chunk(c + 1);
            // drain chunk c (<=2 outstanding belong to chunk c+1)
            asm volatile("s_wait_asynccnt 2" ::: "memory");
        } else {
            asm volatile("s_wait_asynccnt 0" ::: "memory");
        }
        __syncthreads();

        const float* buf = &ldsXF[(c & 1) * XBUF];

        // bucket piece count straight from the f32 tile
        if (s_side == 0 && c < 24) {
            const float* q = &buf[sr * XROW + sj];
            psum += (q[0] + q[1]) + (q[2] + q[3]);
        }

        const int k0 = c * 32;
        const float* arow = &buf[(rt * 16 + m) * XROW];
        const v16bf a = make_afrag_f32(arow + 8 * hh, arow + 16 + 8 * hh);
#pragma unroll
        for (int ct = 0; ct < 8; ++ct) {
            const int n0 = cg * 128 + ct * 16;
            // B-frag: lane = 16h+n holds column n, K = {8h..8h+7} and {16+8h..23+8h}
            const bf16_t* wcol = wT + (size_t)(n0 + m) * KPAD + k0;
            const v16bf b = make_afrag(wcol + 8 * hh, wcol + 16 + 8 * hh);
            acc[ct] = __builtin_amdgcn_wmma_f32_16x16x32_bf16(
                false, a, false, b, (short)0, acc[ct], false, false);
        }
        __syncthreads();   // release buf[c&1] for chunk c+2's asyncs
    }

    // piece count -> bucket (reduce the 8 staging threads of each row)
    psum += __shfl_xor(psum, 1);
    psum += __shfl_xor(psum, 2);
    psum += __shfl_xor(psum, 4);
    if (((t & 7) == 0) && s_side == 0) {
        int bi = (int)(psum * (8.0f / 33.0f));
        bi = bi < 0 ? 0 : (bi > 7 ? 7 : bi);
        ldsBk[s_smp] = bi;
    }

    // FT epilogue: bias + crelu -> ldsC (bf16), concat order [stm | nstm]
#pragma unroll
    for (int ct = 0; ct < 8; ++ct) {
        const int n    = cg * 128 + ct * 16 + m;     // C layout: N = lane&15
        const float bn = ft_b[n];
#pragma unroll
        for (int rv = 0; rv < 8; ++rv) {
            const int gr = rt * 16 + rv + 8 * hh;    // C layout: M = rv + 8*half
            const int ss = gr & 15, sd = gr >> 4;
            float v = acc[ct][rv] + bn;
            v = fminf(fmaxf(v, 0.0f), 1.0f);
            ldsC[ss * 1032 + sd * 512 + n] = (bf16_t)v;
        }
    }
    __syncthreads();

    // ---------------- h1: all 8 buckets, (16 x 1024) @ (1024 x 256) ----------
    v8f hacc[2] = {zero8, zero8};
    for (int kc = 0; kc < 32; ++kc) {
        const int k0 = kc * 32;
        const bf16_t* crow = &ldsC[m * 1032 + k0];
        const v16bf a = make_afrag(crow + 8 * hh, crow + 16 + 8 * hh);
#pragma unroll
        for (int i = 0; i < 2; ++i) {
            const int n0 = (w + 8 * i) * 16;
            const bf16_t* hcol = h1t + (size_t)(n0 + m) * COMB + k0;
            const v16bf b = make_afrag(hcol + 8 * hh, hcol + 16 + 8 * hh);
            hacc[i] = __builtin_amdgcn_wmma_f32_16x16x32_bf16(
                false, a, false, b, (short)0, hacc[i], false, false);
        }
    }
#pragma unroll
    for (int i = 0; i < 2; ++i) {
        const int n    = (w + 8 * i) * 16 + m;
        const float bn = h1_b[n];
#pragma unroll
        for (int rv = 0; rv < 8; ++rv) {
            const int ss = rv + 8 * hh;
            float v = hacc[i][rv] + bn;
            v = fminf(fmaxf(v, 0.0f), 1.0f);
            ldsH[ss * 264 + n] = (bf16_t)v;
        }
    }
    __syncthreads();

    // ---------------- h2: per bucket, (16 x 32) @ (32 x 32), K=32 ------------
#pragma unroll
    for (int i = 0; i < 2; ++i) {
        const int t2 = w + 8 * i;          // 0..15
        const int bk = t2 >> 1;
        const int p0 = (t2 & 1) * 16;
        const bf16_t* hrow = &ldsH[m * 264 + bk * 32];
        const v16bf a = make_afrag(hrow + 8 * hh, hrow + 16 + 8 * hh);
        const bf16_t* h2col = h2t + (size_t)bk * 1024 + (p0 + m) * 32;
        const v16bf b = make_afrag(h2col + 8 * hh, h2col + 16 + 8 * hh);
        v8f g = __builtin_amdgcn_wmma_f32_16x16x32_bf16(
            false, a, false, b, (short)0, zero8, false, false);
        const int n    = bk * 32 + p0 + m;
        const float bn = h2_b[n];
#pragma unroll
        for (int rv = 0; rv < 8; ++rv) {
            const int ss = rv + 8 * hh;
            float v = g[rv] + bn;
            v = fminf(fmaxf(v, 0.0f), 1.0f);
            ldsH2[ss * 264 + n] = (bf16_t)v;
        }
    }
    __syncthreads();

    // ---------------- h3 + bucket select -------------------------------------
    if (t < 16) {
        const int bk = ldsBk[t];
        const bf16_t* hr = &ldsH2[t * 264 + bk * 32];
        float sum = h3_b[bk];
#pragma unroll
        for (int o = 0; o < 32; ++o) sum += (float)hr[o] * h3_w[bk * 32 + o];
        out[b0 + t] = sum;
    }
}

extern "C" void kernel_launch(void* const* d_in, const int* in_sizes, int n_in,
                              void* d_out, int out_size, void* d_ws, size_t ws_size,
                              hipStream_t stream) {
    const float* x    = (const float*)d_in[0];
    const float* ft_w = (const float*)d_in[1];
    const float* ft_b = (const float*)d_in[2];
    const float* h1_w = (const float*)d_in[3];
    const float* h1_b = (const float*)d_in[4];
    const float* h2_w = (const float*)d_in[5];
    const float* h2_b = (const float*)d_in[6];
    const float* h3_w = (const float*)d_in[7];
    const float* h3_b = (const float*)d_in[8];
    float* out = (float*)d_out;

    const int B = in_sizes[0] / (2 * IN_FEATS);

    bf16_t* wT  = (bf16_t*)((char*)d_ws + WT_OFF);
    bf16_t* h1t = (bf16_t*)((char*)d_ws + H1T_OFF);
    bf16_t* h2t = (bf16_t*)((char*)d_ws + H2T_OFF);

    convert_weights<<<(KPAD * FT + 255) / 256, 256, 0, stream>>>(
        ft_w, h1_w, h2_w, wT, h1t, h2t);

    nnue_fused<<<B / 16, 256, 0, stream>>>(
        x, ft_b, h1_b, h2_b, h3_w, h3_b, wT, h1t, h2t, out);
}
